// Physics_Attention_Irregular_Mesh_8272107012823
// MI455X (gfx1250) — matrix operand order repaired and checked by reading, not compiled, not run on hardware
//
#include <hip/hip_runtime.h>
#include <hip/hip_bf16.h>
#include <math.h>
#include <stdint.h>

typedef __bf16 bf16t;
typedef __attribute__((ext_vector_type(16))) __bf16 v16bf;
typedef __attribute__((ext_vector_type(8)))  float  v8f;

#define WMMA_BF16(a,b,c) __builtin_amdgcn_wmma_f32_16x16x32_bf16(false,(a),false,(b),(short)0,(c),false,false)

// ---- problem constants ----
static const int cG = 32;            // B*H
static const int cGPM = 66, cPAD = 96, cNC = 48;

__device__ inline float wred(float v){
  #pragma unroll
  for (int o = 16; o; o >>= 1) v += __shfl_xor(v, o);
  return v;
}

// async 16B global->LDS copy (CDNA5 ASYNC path), per-lane addresses
__device__ inline void async_g2l_b128(void* lds, const void* gptr){
  unsigned int loff = (unsigned int)(uintptr_t)lds;          // LDS aperture: low 32 bits = LDS offset
  unsigned long long ga = (unsigned long long)(uintptr_t)gptr;
  asm volatile("global_load_async_to_lds_b128 %0, %1, off" :: "v"(loff), "v"(ga) : "memory");
}
__device__ inline void async_wait0(){
  asm volatile("s_wait_asynccnt 0x0" ::: "memory");
}

// =====================================================================
// x (f32) -> xb (bf16) once; removes cvts + halves bytes in GEMM1 loop
// =====================================================================
__global__ __launch_bounds__(256) void k_cvtx(const float* __restrict__ x,
                                              bf16t* __restrict__ xb, long n){
  long i = ((long)blockIdx.x*256 + threadIdx.x)*8;
  if (i < n){
    #pragma unroll
    for (int j = 0; j < 8; j++) xb[i+j] = (bf16t)x[i+j];
  }
}

// =====================================================================
// weight prep: transpose big weights to bf16 [out][in]
// =====================================================================
__global__ void k_prepw(const float* __restrict__ Win, const float* __restrict__ Wout,
                        const float* __restrict__ Wsl,
                        bf16t* __restrict__ Win_t, bf16t* __restrict__ Wout_t,
                        bf16t* __restrict__ Asl){
  int t = blockIdx.x*256 + threadIdx.x;
  if (t < 131072){ int o = t >> 8, k = t & 255; Win_t[t] = (bf16t)Win[k*512 + o]; }
  else if (t < 262144){ int u = t - 131072; int c = u >> 9, k = u & 511; Wout_t[u] = (bf16t)Wout[k*256 + c]; }
  else if (t < 266240){ int u = t - 262144; int s = u >> 6, d = u & 63; Asl[u] = (bf16t)Wsl[d*64 + s]; }
}

// =====================================================================
// GEMM1 (transposed): xp_tr[(b*8+h)*64+d][n] = (x @ Win + b)^T
// B tile (64 tokens x 32 k, bf16) staged in LDS via async copies and
// shared by all 8 waves (8 channel tiles) of the workgroup.
// =====================================================================
__global__ __launch_bounds__(256) void k_gemm_inx(const bf16t* __restrict__ xb,
    const bf16t* __restrict__ Wt, const float* __restrict__ bias,
    bf16t* __restrict__ xp_tr){
  __shared__ bf16t sB[64*32];                 // [token][k], 4KB
  int tid = threadIdx.x;
  int wave = tid >> 5, lane = tid & 31, lm = lane & 15, lh = lane >> 4;
  int wg = blockIdx.x;
  int cgrp = wg & 3;                          // 4 groups x 128 channels
  long t0 = (long)(wg >> 2)*64;               // token group of 64
  int mt = cgrp*8 + wave;                     // channel tile (16 ch)
  int c0 = mt*16;
  int stok = tid >> 2, spart = tid & 3;       // staging: 1x16B per thread
  v8f acc[4] = {};
  for (int k0 = 0; k0 < 256; k0 += 32){
    __syncthreads();
    async_g2l_b128(&sB[stok*32 + spart*8],
                   xb + (t0 + stok)*256 + k0 + spart*8);
    async_wait0();
    __syncthreads();
    v16bf a;
    {
      const bf16t* ar = Wt + (c0 + lm)*256 + k0;
      #pragma unroll
      for (int j = 0; j < 16; j++) a[j] = ar[lh*16 + j];
    }
    #pragma unroll
    for (int nt = 0; nt < 4; nt++){
      const bf16t* br = sB + (nt*16 + lm)*32 + lh*16;
      v16bf bb;
      #pragma unroll
      for (int j = 0; j < 16; j++) bb[j] = br[j];
      acc[nt] = WMMA_BF16(a, bb, acc[nt]);
    }
  }
  #pragma unroll
  for (int nt = 0; nt < 4; nt++){
    #pragma unroll
    for (int r = 0; r < 8; r++){
      int c = c0 + r + lh*8;
      long t = t0 + nt*16 + lm;
      int b = (int)(t >> 14), n = (int)(t & 16383);
      int h = c >> 6, d = c & 63;
      xp_tr[((long)((b*8 + h)*64 + d))*16384 + n] = (bf16t)(acc[nt][r] + bias[c]);
    }
  }
}

// =====================================================================
// logits (transposed): logt[bh*64+s][n] = sum_d xp[n,d]*Wsl[d,s] + bsl[s]
// =====================================================================
__global__ __launch_bounds__(256) void k_logits(const bf16t* __restrict__ Asl,
    const float* __restrict__ bsl, const bf16t* __restrict__ xp_tr,
    bf16t* __restrict__ logt){
  int w = (blockIdx.x*256 + threadIdx.x) >> 5;
  int lane = threadIdx.x & 31, lm = lane & 15, lh = lane >> 4;
  int bh = w >> 10; long n0 = (long)(w & 1023) * 16;
  v8f acc[4] = {};
  for (int k0 = 0; k0 < 64; k0 += 32){
    v16bf bb;
    #pragma unroll
    for (int j = 0; j < 16; j++){
      int d = k0 + j + lh*16;
      bb[j] = xp_tr[((long)(bh*64 + d))*16384 + n0 + lm];
    }
    #pragma unroll
    for (int st = 0; st < 4; st++){
      const bf16t* ar = Asl + (st*16 + lm)*64 + k0;
      v16bf a;
      #pragma unroll
      for (int j = 0; j < 16; j++) a[j] = ar[lh*16 + j];
      acc[st] = WMMA_BF16(a, bb, acc[st]);
    }
  }
  #pragma unroll
  for (int st = 0; st < 4; st++)
    #pragma unroll
    for (int r = 0; r < 8; r++){
      int s = st*16 + r + lh*8;
      logt[((long)(bh*64 + s))*16384 + n0 + lm] = (bf16t)(acc[st][r] + bsl[s]);
    }
}

// =====================================================================
// per-token: LN(ln_t) -> ada -> tau -> softmax(logits/tau) -> sw_tr (bf16)
// =====================================================================
__global__ __launch_bounds__(256) void k_softmax(const bf16t* __restrict__ xp_tr,
    const bf16t* __restrict__ logt, const float* __restrict__ lng,
    const float* __restrict__ lnb, const float* __restrict__ adaW,
    const float* __restrict__ adab, bf16t* __restrict__ sw_tr){
  long idx = (long)blockIdx.x*256 + threadIdx.x;
  int bh = (int)(idx >> 14); int n = (int)(idx & 16383);
  const bf16t* col = xp_tr + (long)bh*64*16384 + n;
  float xv[64]; float mean = 0.f;
  #pragma unroll
  for (int d = 0; d < 64; d++){ xv[d] = (float)col[(long)d*16384]; mean += xv[d]; }
  mean *= (1.f/64.f);
  float var = 0.f;
  #pragma unroll
  for (int d = 0; d < 64; d++){ float t = xv[d]-mean; var += t*t; }
  var *= (1.f/64.f);
  float rs = rsqrtf(var + 1e-5f);
  float dot = adab[0];
  #pragma unroll
  for (int d = 0; d < 64; d++) dot += ((xv[d]-mean)*rs*lng[d] + lnb[d]) * adaW[d];
  float tau = 0.5f + dot; tau = fminf(fmaxf(tau, 0.1f), 2.0f);
  float it = 1.f / tau;
  const bf16t* lc = logt + (long)bh*64*16384 + n;
  float lg[64]; float mx = -1e30f;
  #pragma unroll
  for (int s = 0; s < 64; s++){ lg[s] = (float)lc[(long)s*16384] * it; mx = fmaxf(mx, lg[s]); }
  float sum = 0.f;
  #pragma unroll
  for (int s = 0; s < 64; s++){ lg[s] = __expf(lg[s]-mx); sum += lg[s]; }
  float inv = 1.f / sum;
  bf16t* oc = sw_tr + (long)bh*64*16384 + n;
  #pragma unroll
  for (int s = 0; s < 64; s++) oc[(long)s*16384] = (bf16t)(lg[s]*inv);
}

// =====================================================================
// eid[bh][s][d] = sum_n sw[n,s]*xp[n,d]   (K = 16384, one WG per bh)
// =====================================================================
__global__ __launch_bounds__(256) void k_eid(const bf16t* __restrict__ sw_tr,
    const bf16t* __restrict__ xp_tr, float* __restrict__ eid){
  int bh = blockIdx.x;
  int wave = threadIdx.x >> 5, lane = threadIdx.x & 31;
  int lm = lane & 15, lh = lane >> 4;
  for (int ti = wave; ti < 16; ti += 8){
    int st = ti >> 2, dt = ti & 3;
    v8f acc = {};
    for (long n0 = 0; n0 < 16384; n0 += 32){
      v16bf a, bb;
      const bf16t* ar = sw_tr + ((long)(bh*64 + st*16 + lm))*16384 + n0;
      const bf16t* br = xp_tr + ((long)(bh*64 + dt*16 + lm))*16384 + n0 + lh*16;
      __builtin_prefetch(ar + 256, 0, 3);     // global_prefetch_b8, stream-ahead
      __builtin_prefetch(br + 256, 0, 3);
      #pragma unroll
      for (int j = 0; j < 16; j++) a[j] = ar[lh*16 + j];
      #pragma unroll
      for (int j = 0; j < 16; j++) bb[j] = br[j];
      acc = WMMA_BF16(a, bb, acc);
    }
    #pragma unroll
    for (int r = 0; r < 8; r++)
      eid[((long)(bh*64) + st*16 + r + lh*8)*64 + dt*16 + lm] = acc[r];
  }
}

// norm[bh][s] = sum_n sw; eidp raw sums with pos
__global__ __launch_bounds__(256) void k_normp(const bf16t* __restrict__ sw_tr,
    const float* __restrict__ pos, float* __restrict__ norm, float* __restrict__ eidp){
  int w = (blockIdx.x*256 + threadIdx.x) >> 5; if (w >= cG*64) return;
  int lane = threadIdx.x & 31;
  int bh = w >> 6, b = bh >> 3;
  const bf16t* row = sw_tr + ((long)w)*16384;
  const float* pb = pos + (long)b*16384*3;
  float s0=0,s1=0,s2=0,s3=0;
  for (int n = lane; n < 16384; n += 32){
    float v = (float)row[n];
    s0 += v; s1 += v*pb[n*3]; s2 += v*pb[n*3+1]; s3 += v*pb[n*3+2];
  }
  s0 = wred(s0); s1 = wred(s1); s2 = wred(s2); s3 = wred(s3);
  if (lane == 0){ norm[w]=s0; eidp[w*3]=s1; eidp[w*3+1]=s2; eidp[w*3+2]=s3; }
}

// build padded erwin inputs x96/p96 (divide by norm+0.001, concat mem, pad 96)
__global__ void k_build96(const float* __restrict__ eid, const float* __restrict__ eidp,
    const float* __restrict__ norm, const float* __restrict__ mstate,
    const float* __restrict__ mpos, float* __restrict__ x96, float* __restrict__ p96){
  int t = blockIdx.x*blockDim.x + threadIdx.x; if (t >= cG*cPAD) return;
  int g = t / cPAD, i = t % cPAD, j = i % cGPM, h = g & 7;
  if (j < 64){
    float nm = 1.f/(norm[g*64 + j] + 0.001f);
    for (int d = 0; d < 64; d++) x96[(long)t*64 + d] = eid[((long)g*64 + j)*64 + d]*nm;
    for (int c = 0; c < 3; c++)  p96[t*3 + c] = eidp[(g*64 + j)*3 + c]*nm;
  } else {
    int m = j - 64;
    for (int d = 0; d < 64; d++) x96[(long)t*64 + d] = mstate[(h*2 + m)*64 + d];
    for (int c = 0; c < 3; c++)  p96[t*3 + c] = mpos[(h*2 + m)*3 + c];
  }
}

// =====================================================================
// generic guarded WMMA GEMM: out[M][N] = A[M][K]@W[K][N] (+bias)(+res)(scatter)
// =====================================================================
__global__ __launch_bounds__(256) void k_gemm_small(const float* __restrict__ A,
    const float* __restrict__ W, const float* __restrict__ bias,
    float* __restrict__ out, const float* __restrict__ res,
    const int* __restrict__ scat, int scatP, int M, int N, int K){
  int w = (blockIdx.x*256 + threadIdx.x) >> 5;
  int lane = threadIdx.x & 31, lm = lane & 15, lh = lane >> 4;
  int ntiles = (N + 15) >> 4, mtiles = (M + 15) >> 4;
  if (w >= mtiles*ntiles) return;
  int mt = w / ntiles, nt = w % ntiles;
  v8f acc = {};
  for (int k0 = 0; k0 < K; k0 += 32){
    v16bf a, bb;
    int m = mt*16 + lm;
    #pragma unroll
    for (int j = 0; j < 16; j++){
      int k = k0 + j + lh*16;
      a[j] = (m < M && k < K) ? (bf16t)A[(long)m*K + k] : (bf16t)0.f;
    }
    int n = nt*16 + lm;
    #pragma unroll
    for (int j = 0; j < 16; j++){
      int k = k0 + j + lh*16;
      bb[j] = (n < N && k < K) ? (bf16t)W[(long)k*N + n] : (bf16t)0.f;
    }
    acc = WMMA_BF16(a, bb, acc);
  }
  int n = nt*16 + lm;
  if (n < N){
    #pragma unroll
    for (int r = 0; r < 8; r++){
      int m = mt*16 + r + lh*8;
      if (m < M){
        float v = acc[r];
        if (bias) v += bias[n];
        if (res)  v += res[(long)m*N + n];
        long mo = m;
        if (scat) mo = (long)(m / scatP)*scatP + scat[m];
        out[mo*N + n] = v;
      }
    }
  }
}

// =====================================================================
// morton order (bitonic argsort of interleaved codes); grid.y: 0 plain 1 rotated
// =====================================================================
template<int P, int PADP>
__global__ void k_morton(const float* __restrict__ pos, int* __restrict__ ordA,
    int* __restrict__ invA, int* __restrict__ ordB, int* __restrict__ invB){
  __shared__ float sp[P*3];
  __shared__ unsigned long long key[PADP];
  __shared__ float mnx[3], mxx[3];
  int g = blockIdx.x, i = threadIdx.x, rot = blockIdx.y;
  const float R00=0.5f, R01=-0.70710678f, R02=0.5f,
              R10=0.5f, R11= 0.70710678f, R12=0.5f,
              R20=-0.70710678f, R21=0.f,  R22=0.70710678f;
  if (i < P){
    float p0=pos[(g*P+i)*3], p1=pos[(g*P+i)*3+1], p2=pos[(g*P+i)*3+2];
    float q0=p0,q1=p1,q2=p2;
    if (rot){ q0=R00*p0+R01*p1+R02*p2; q1=R10*p0+R11*p1+R12*p2; q2=R20*p0+R21*p1+R22*p2; }
    sp[i*3]=q0; sp[i*3+1]=q1; sp[i*3+2]=q2;
  }
  __syncthreads();
  if (i == 0){
    for (int d = 0; d < 3; d++){
      float mn = sp[d], mx = sp[d];
      for (int t = 1; t < P; t++){ float v = sp[t*3+d]; mn = fminf(mn,v); mx = fmaxf(mx,v); }
      mnx[d]=mn; mxx[d]=mx;
    }
  }
  __syncthreads();
  unsigned long long kk = ~0ull;
  if (i < P){
    int code = 0, q[3];
    for (int d = 0; d < 3; d++){
      float v = (sp[i*3+d]-mnx[d])/(mxx[d]-mnx[d]+1e-6f)*1023.f;
      v = fminf(fmaxf(v,0.f),1023.f); q[d] = (int)v;
    }
    for (int bit = 0; bit < 10; bit++)
      for (int d = 0; d < 3; d++) code |= ((q[d]>>bit)&1) << (3*bit + d);
    kk = ((unsigned long long)(unsigned)code << 32) | (unsigned)i;
  }
  key[i] = kk;
  __syncthreads();
  for (int k = 2; k <= PADP; k <<= 1)
    for (int j = k >> 1; j > 0; j >>= 1){
      int ix = i ^ j;
      if (ix > i){
        bool up = ((i & k) == 0);
        unsigned long long a = key[i], b = key[ix];
        if ((a > b) == up){ key[i] = b; key[ix] = a; }
      }
      __syncthreads();
    }
  if (i < P){
    int oi = (int)(key[i] & 0xffffffffu);
    int* ord = rot ? ordB : ordA; int* inv = rot ? invB : invA;
    ord[g*P + i] = oi; inv[g*P + oi] = i;
  }
}

// gather positions by order + per-ball means
__global__ void k_gpos(const float* __restrict__ pos, const int* __restrict__ ord,
                       float* __restrict__ pog, float* __restrict__ bmean, int P, int bs){
  extern __shared__ float sp[];
  int g = blockIdx.x, i = threadIdx.x;
  if (i < P){
    int s = ord[g*P + i];
    for (int c = 0; c < 3; c++){ float v = pos[(g*P+s)*3 + c]; sp[i*3+c] = v; pog[(g*P+i)*3+c] = v; }
  }
  __syncthreads();
  int nb = P / bs;
  if (i < nb){
    for (int c = 0; c < 3; c++){
      float s = 0.f;
      for (int t = 0; t < bs; t++) s += sp[(i*bs+t)*3 + c];
      bmean[(g*nb+i)*3 + c] = s/(float)bs;
    }
  }
}

// gather x by order, save residual copy, LN1, add pe(rel)
template<int DIM>
__global__ __launch_bounds__(256) void k_prep_block(const float* __restrict__ x,
    const float* __restrict__ pog, const float* __restrict__ bmean,
    const int* __restrict__ ord, const float* __restrict__ lng, const float* __restrict__ lnb,
    const float* __restrict__ peW, const float* __restrict__ peb,
    float* __restrict__ xg, float* __restrict__ y, int P, int bs, int rows){
  int w = (blockIdx.x*256 + threadIdx.x) >> 5; if (w >= rows) return;
  int lane = threadIdx.x & 31;
  int g = w / P, i = w % P;
  int src = g*P + ord[w];
  const int E = DIM/32;
  float v[E]; float s = 0.f;
  #pragma unroll
  for (int e = 0; e < E; e++){
    v[e] = x[(long)src*DIM + e*32 + lane];
    xg[(long)w*DIM + e*32 + lane] = v[e];
    s += v[e];
  }
  s = wred(s); float mean = s/(float)DIM;
  float q = 0.f;
  #pragma unroll
  for (int e = 0; e < E; e++){ float t = v[e]-mean; q += t*t; }
  q = wred(q); float rs = rsqrtf(q/(float)DIM + 1e-5f);
  int nb = P/bs, ball = i/bs;
  float r0 = pog[w*3]   - bmean[(g*nb+ball)*3];
  float r1 = pog[w*3+1] - bmean[(g*nb+ball)*3+1];
  float r2 = pog[w*3+2] - bmean[(g*nb+ball)*3+2];
  #pragma unroll
  for (int e = 0; e < E; e++){
    int d = e*32 + lane;
    float o = (v[e]-mean)*rs*lng[d] + lnb[d];
    o += r0*peW[d] + r1*peW[DIM + d] + r2*peW[2*DIM + d] + peb[d];
    y[(long)w*DIM + d] = o;
  }
}

template<int DIM>
__global__ __launch_bounds__(256) void k_ln(const float* __restrict__ x,
    const float* __restrict__ g, const float* __restrict__ b,
    float* __restrict__ out, int rows){
  int w = (blockIdx.x*256 + threadIdx.x) >> 5; if (w >= rows) return;
  int lane = threadIdx.x & 31;
  const int E = DIM/32;
  float v[E]; float s = 0.f;
  #pragma unroll
  for (int e = 0; e < E; e++){ v[e] = x[(long)w*DIM + e*32 + lane]; s += v[e]; }
  s = wred(s); float mean = s/(float)DIM;
  float q = 0.f;
  #pragma unroll
  for (int e = 0; e < E; e++){ float t = v[e]-mean; q += t*t; }
  q = wred(q); float rs = rsqrtf(q/(float)DIM + 1e-5f);
  #pragma unroll
  for (int e = 0; e < E; e++){
    int d = e*32 + lane;
    out[(long)w*DIM + d] = (v[e]-mean)*rs*g[d] + b[d];
  }
}

// ball attention with distance bias (small, VALU)
template<int DIM, int HEADS, int BS>
__global__ __launch_bounds__(BS*HEADS) void k_attn(const float* __restrict__ qkv,
    const float* __restrict__ pog, const float* __restrict__ sigma,
    float* __restrict__ att_o){
  const int DHD = DIM/HEADS;
  __shared__ float sk[BS*DIM], sv[BS*DIM], sp[BS*3];
  int row0 = blockIdx.x * BS;
  int tid = threadIdx.x; int i = tid % BS, h = tid / BS;
  for (int idx = tid; idx < BS*DIM; idx += BS*HEADS){
    int ii = idx / DIM, c = idx % DIM;
    sk[idx] = qkv[(long)(row0+ii)*3*DIM + DIM   + c];
    sv[idx] = qkv[(long)(row0+ii)*3*DIM + 2*DIM + c];
  }
  if (tid < BS*3) sp[tid] = pog[row0*3 + tid];
  __syncthreads();
  float q[DHD];
  #pragma unroll
  for (int d = 0; d < DHD; d++) q[d] = qkv[(long)(row0+i)*3*DIM + h*DHD + d];
  float sg = sigma[h];
  float p0 = sp[i*3], p1 = sp[i*3+1], p2 = sp[i*3+2];
  const float scl = rsqrtf((float)DHD);
  float lo[BS]; float mx = -1e30f;
  #pragma unroll
  for (int j = 0; j < BS; j++){
    float dot = 0.f;
    #pragma unroll
    for (int d = 0; d < DHD; d++) dot += q[d]*sk[j*DIM + h*DHD + d];
    float dx = p0-sp[j*3], dy = p1-sp[j*3+1], dz = p2-sp[j*3+2];
    float dist = sqrtf(dx*dx + dy*dy + dz*dz + 1e-9f);
    float l = dot*scl + sg*dist; lo[j] = l; mx = fmaxf(mx, l);
  }
  float sum = 0.f;
  #pragma unroll
  for (int j = 0; j < BS; j++){ lo[j] = __expf(lo[j]-mx); sum += lo[j]; }
  float inv = 1.f/sum;
  float o[DHD];
  #pragma unroll
  for (int d = 0; d < DHD; d++) o[d] = 0.f;
  #pragma unroll
  for (int j = 0; j < BS; j++){
    float wt = lo[j]*inv;
    #pragma unroll
    for (int d = 0; d < DHD; d++) o[d] += wt*sv[j*DIM + h*DHD + d];
  }
  #pragma unroll
  for (int d = 0; d < DHD; d++) att_o[(long)(row0+i)*DIM + h*DHD + d] = o[d];
}

__global__ void k_swiglu(const float* __restrict__ h1, const float* __restrict__ h2,
                         float* __restrict__ hm, long n){
  long i = (long)blockIdx.x*256 + threadIdx.x;
  if (i < n){ float a = h1[i]; hm[i] = (a/(1.f + __expf(-a)))*h2[i]; }
}

// pool: build [xs pair | rel] concat rows, cpos, rel6
__global__ void k_pool_prep(const float* __restrict__ ex, const float* __restrict__ p96,
    const int* __restrict__ ord0, float* __restrict__ cat, float* __restrict__ cpos,
    float* __restrict__ rel6){
  int t = blockIdx.x*256 + threadIdx.x; if (t >= cG*cNC) return;
  int g = t / cNC, c = t % cNC;
  int j0 = ord0[g*96 + 2*c], j1 = ord0[g*96 + 2*c + 1];
  float a0[3], a1[3];
  for (int k = 0; k < 3; k++){
    a0[k] = p96[(g*96 + j0)*3 + k]; a1[k] = p96[(g*96 + j1)*3 + k];
    float cp = 0.5f*(a0[k] + a1[k]);
    cpos[t*3 + k] = cp;
    rel6[t*6 + k]     = a0[k] - cp;
    rel6[t*6 + 3 + k] = a1[k] - cp;
  }
  float* row = cat + (long)t*134;
  for (int d = 0; d < 64; d++){
    row[d]      = ex[((long)g*96 + j0)*64 + d];
    row[64 + d] = ex[((long)g*96 + j1)*64 + d];
  }
  for (int k = 0; k < 6; k++) row[128 + k] = rel6[t*6 + k];
}

__global__ void k_unpool_prep(const float* __restrict__ xc, const float* __restrict__ rel6,
                              float* __restrict__ cat){
  int t = blockIdx.x*256 + threadIdx.x; if (t >= cG*cNC) return;
  float* row = cat + (long)t*134;
  for (int d = 0; d < 128; d++) row[d] = xc[(long)t*128 + d];
  for (int k = 0; k < 6; k++) row[128 + k] = rel6[t*6 + k];
}

// x = LN_unpool(skip + unpool_reshaped[inv0])
__global__ __launch_bounds__(256) void k_unpool_combine(const float* __restrict__ skip,
    const float* __restrict__ unp, const int* __restrict__ inv0,
    const float* __restrict__ lng, const float* __restrict__ lnb, float* __restrict__ ex){
  int w = (blockIdx.x*256 + threadIdx.x) >> 5; if (w >= cG*96) return;
  int lane = threadIdx.x & 31;
  int g = w / 96, j = w % 96;
  int iv = inv0[g*96 + j];
  float v[2]; float s = 0.f;
  #pragma unroll
  for (int e = 0; e < 2; e++){
    int d = e*32 + lane;
    v[e] = skip[(long)w*64 + d] + unp[((long)g*48 + (iv >> 1))*128 + (iv & 1)*64 + d];
    s += v[e];
  }
  s = wred(s); float mean = s*(1.f/64.f);
  float q = 0.f;
  #pragma unroll
  for (int e = 0; e < 2; e++){ float t = v[e]-mean; q += t*t; }
  q = wred(q); float rs = rsqrtf(q*(1.f/64.f) + 1e-5f);
  #pragma unroll
  for (int e = 0; e < 2; e++){
    int d = e*32 + lane;
    ex[(long)w*64 + d] = (v[e]-mean)*rs*lng[d] + lnb[d];
  }
}

// =====================================================================
// out einsum: out_hd[b*16384+n][h*64+d] = sum_s sw[n,s]*upd[s,d]
// =====================================================================
__global__ __launch_bounds__(256) void k_outein(const bf16t* __restrict__ sw_tr,
    const float* __restrict__ ex, bf16t* __restrict__ out_hd){
  int w = (blockIdx.x*256 + threadIdx.x) >> 5;
  int lane = threadIdx.x & 31, lm = lane & 15, lh = lane >> 4;
  int bh = w >> 10; long n0 = (long)(w & 1023)*16;
  v8f acc[4] = {};
  for (int k0 = 0; k0 < 64; k0 += 32){
    v16bf a;
    #pragma unroll
    for (int j = 0; j < 16; j++){
      int s = k0 + j + lh*16;
      a[j] = sw_tr[((long)(bh*64 + s))*16384 + n0 + lm];
    }
    #pragma unroll
    for (int dt = 0; dt < 4; dt++){
      v16bf bb;
      #pragma unroll
      for (int j = 0; j < 16; j++){
        int s = k0 + j + lh*16;
        bb[j] = (bf16t)ex[((long)bh*96 + s)*64 + dt*16 + lm];
      }
      acc[dt] = WMMA_BF16(a, bb, acc[dt]);
    }
  }
  int b = bh >> 3, h = bh & 7;
  #pragma unroll
  for (int dt = 0; dt < 4; dt++)
    #pragma unroll
    for (int r = 0; r < 8; r++){
      long tok = (long)b*16384 + n0 + r + lh*8;
      out_hd[tok*512 + h*64 + dt*16 + lm] = (bf16t)acc[dt][r];
    }
}

// =====================================================================
// final: out = out_hd @ Wout + bout; Wout tile staged in LDS via async copies
// WG = 8 waves x 8 m-tiles sharing one 64-col B tile per K step
// =====================================================================
__global__ __launch_bounds__(256) void k_gemm_out(const bf16t* __restrict__ A,
    const bf16t* __restrict__ Wt, const float* __restrict__ bias, float* __restrict__ out){
  __shared__ bf16t sW[64*32];                 // [col][k], 4KB
  int tid = threadIdx.x;
  int wave = tid >> 5, lane = tid & 31, lm = lane & 15, lh = lane >> 4;
  int wg = blockIdx.x;
  int ct = wg & 3;                            // 4 col groups of 64
  int mt = (wg >> 2)*8 + wave;                // token tile (16 rows)
  int scol = tid >> 2, spart = tid & 3;       // staging map
  v8f acc[4] = {};
  for (int k0 = 0; k0 < 512; k0 += 32){
    __syncthreads();
    async_g2l_b128(&sW[scol*32 + spart*8],
                   Wt + (long)(ct*64 + scol)*512 + k0 + spart*8);
    async_wait0();
    __syncthreads();
    v16bf a;
    {
      const bf16t* ar = A + (long)(mt*16 + lm)*512 + k0;
      #pragma unroll
      for (int j = 0; j < 16; j++) a[j] = ar[lh*16 + j];
    }
    #pragma unroll
    for (int nt = 0; nt < 4; nt++){
      const bf16t* br = sW + (nt*16 + lm)*32 + lh*16;
      v16bf bb;
      #pragma unroll
      for (int j = 0; j < 16; j++) bb[j] = br[j];
      acc[nt] = WMMA_BF16(a, bb, acc[nt]);
    }
  }
  #pragma unroll
  for (int nt = 0; nt < 4; nt++)
    #pragma unroll
    for (int r = 0; r < 8; r++){
      int c = ct*64 + nt*16 + lm;
      out[(long)(mt*16 + r + lh*8)*256 + c] = acc[nt][r] + bias[c];
    }
}

// =====================================================================
// host
// =====================================================================
extern "C" void kernel_launch(void* const* d_in, const int* in_sizes, int n_in,
                              void* d_out, int out_size, void* d_ws, size_t ws_size,
                              hipStream_t stream){
  (void)in_sizes; (void)n_in; (void)out_size; (void)ws_size;
  auto F = [&](int i){ return (const float*)d_in[i]; };

  char* base = (char*)d_ws; size_t off = 0;
  auto alloc = [&](size_t bytes)->void*{
    void* p = base + off; off = (off + bytes + 255) & ~(size_t)255; return p;
  };

  bf16t* xb    = (bf16t*)alloc((size_t)65536*256*2);
  bf16t* xp_tr = (bf16t*)alloc((size_t)cG*64*16384*2);
  bf16t* logt  = (bf16t*)alloc((size_t)cG*64*16384*2);   // reused as out_hd
  bf16t* sw_tr = (bf16t*)alloc((size_t)cG*64*16384*2);
  bf16t* Win_t = (bf16t*)alloc((size_t)512*256*2);
  bf16t* Wout_t= (bf16t*)alloc((size_t)256*512*2);
  bf16t* Asl   = (bf16t*)alloc((size_t)64*64*2);
  float* norm  = (float*)alloc((size_t)cG*64*4);
  float* eid   = (float*)alloc((size_t)cG*64*64*4);
  float* eidp  = (float*)alloc((size_t)cG*64*3*4);
  float* x96   = (float*)alloc((size_t)cG*96*64*4);
  float* p96   = (float*)alloc((size_t)cG*96*3*4);
  float* ex    = (float*)alloc((size_t)cG*96*64*4);
  float* skip  = (float*)alloc((size_t)cG*96*64*4);
  float* xg    = (float*)alloc((size_t)cG*96*64*4);   // also level1 (48*128)
  float* ybuf  = (float*)alloc((size_t)cG*48*128*4);
  float* qkvb  = (float*)alloc((size_t)cG*96*192*4);  // also level1 (48*384)
  float* atto  = (float*)alloc((size_t)cG*96*64*4);
  float* ln2b  = (float*)alloc((size_t)cG*96*64*4);
  float* h1    = (float*)alloc((size_t)cG*96*256*4);  // also level1 (48*512)
  float* h2    = (float*)alloc((size_t)cG*96*256*4);
  float* hm    = (float*)alloc((size_t)cG*96*256*4);
  int* ord0 = (int*)alloc((size_t)cG*96*4);  int* inv0 = (int*)alloc((size_t)cG*96*4);
  int* rot0 = (int*)alloc((size_t)cG*96*4);  int* irot0= (int*)alloc((size_t)cG*96*4);
  int* ord1 = (int*)alloc((size_t)cG*48*4);  int* inv1 = (int*)alloc((size_t)cG*48*4);
  int* rot1 = (int*)alloc((size_t)cG*48*4);  int* irot1= (int*)alloc((size_t)cG*48*4);
  float* pog   = (float*)alloc((size_t)cG*96*3*4);
  float* bmean = (float*)alloc((size_t)cG*8*3*4);
  float* cpos  = (float*)alloc((size_t)cG*48*3*4);
  float* rel6  = (float*)alloc((size_t)cG*48*6*4);
  float* cat   = (float*)alloc((size_t)cG*48*134*4);
  float* poolo = (float*)alloc((size_t)cG*48*128*4);
  float* xc    = (float*)alloc((size_t)cG*48*128*4);
  float* unp   = (float*)alloc((size_t)cG*48*128*4);
  bf16t* out_hd = logt;

  auto gemmS = [&](const float* A, const float* W, const float* bias, float* out,
                   const float* res, const int* scat, int scatP, int M, int N, int K){
    int waves = ((M+15)/16)*((N+15)/16);
    k_gemm_small<<<(waves+7)/8, 256, 0, stream>>>(A, W, bias, out, res, scat, scatP, M, N, K);
  };

  // ---- stage 1: projections + slice weights ----
  k_cvtx<<<8192, 256, 0, stream>>>(F(0), xb, (long)65536*256);
  k_prepw<<<(266240+255)/256, 256, 0, stream>>>(F(2), F(12), F(4), Win_t, Wout_t, Asl);
  k_gemm_inx<<<4096, 256, 0, stream>>>(xb, Win_t, F(3), xp_tr);
  k_logits<<<4096, 256, 0, stream>>>(Asl, F(5), xp_tr, logt);
  k_softmax<<<2048, 256, 0, stream>>>(xp_tr, logt, F(6), F(7), F(8), F(9), sw_tr);
  k_eid<<<cG, 256, 0, stream>>>(sw_tr, xp_tr, eid);
  k_normp<<<(cG*64)/8, 256, 0, stream>>>(sw_tr, F(1), norm, eidp);
  k_build96<<<(cG*96+255)/256, 256, 0, stream>>>(eid, eidp, norm, F(10), F(11), x96, p96);

  // ---- erwin ----
  gemmS(x96, F(14), F(15), ex, nullptr, nullptr, 0, cG*96, 64, 64);        // embed
  k_morton<96,128><<<dim3(cG,2), 128, 0, stream>>>(p96, ord0, inv0, rot0, irot0);

  auto runBlock0 = [&](int pb, const int* ord){
    int rows = cG*96;
    k_gpos<<<cG, 96, 96*3*4, stream>>>(p96, ord, pog, bmean, 96, 32);
    k_prep_block<64><<<(rows+7)/8, 256, 0, stream>>>(ex, pog, bmean, ord,
        F(pb+0), F(pb+1), F(pb+2), F(pb+3), xg, ybuf, 96, 32, rows);
    gemmS(ybuf, F(pb+4), F(pb+5), qkvb, nullptr, nullptr, 0, rows, 192, 64);
    k_attn<64,4,32><<<cG*3, 128, 0, stream>>>(qkvb, pog, F(pb+6), atto);
    gemmS(atto, F(pb+7), F(pb+8), xg, xg, nullptr, 0, rows, 64, 64);
    k_ln<64><<<(rows+7)/8, 256, 0, stream>>>(xg, F(pb+9), F(pb+10), ln2b, rows);
    gemmS(ln2b, F(pb+11), F(pb+12), h1, nullptr, nullptr, 0, rows, 256, 64);
    gemmS(ln2b, F(pb+13), F(pb+14), h2, nullptr, nullptr, 0, rows, 256, 64);
    k_swiglu<<<(rows*256+255)/256, 256, 0, stream>>>(h1, h2, hm, (long)rows*256);
    gemmS(hm, F(pb+15), F(pb+16), ex, xg, ord, 96, rows, 64, 256);
  };
  auto runBlock1 = [&](int pb, const int* ord){
    int rows = cG*48;
    k_gpos<<<cG, 48, 48*3*4, stream>>>(cpos, ord, pog, bmean, 48, 16);
    k_prep_block<128><<<(rows+7)/8, 256, 0, stream>>>(xc, pog, bmean, ord,
        F(pb+0), F(pb+1), F(pb+2), F(pb+3), xg, ybuf, 48, 16, rows);
    gemmS(ybuf, F(pb+4), F(pb+5), qkvb, nullptr, nullptr, 0, rows, 384, 128);
    k_attn<128,8,16><<<cG*3, 128, 0, stream>>>(qkvb, pog, F(pb+6), atto);
    gemmS(atto, F(pb+7), F(pb+8), xg, xg, nullptr, 0, rows, 128, 128);
    k_ln<128><<<(rows+7)/8, 256, 0, stream>>>(xg, F(pb+9), F(pb+10), ln2b, rows);
    gemmS(ln2b, F(pb+11), F(pb+12), h1, nullptr, nullptr, 0, rows, 512, 128);
    gemmS(ln2b, F(pb+13), F(pb+14), h2, nullptr, nullptr, 0, rows, 512, 128);
    k_swiglu<<<(rows*512+255)/256, 256, 0, stream>>>(h1, h2, hm, (long)rows*512);
    gemmS(hm, F(pb+15), F(pb+16), xc, xg, ord, 48, rows, 128, 512);
  };

  runBlock0(16, ord0);
  runBlock0(33, rot0);
  hipMemcpyAsync(skip, ex, (size_t)cG*96*64*4, hipMemcpyDeviceToDevice, stream);
  k_pool_prep<<<(cG*cNC+255)/256, 256, 0, stream>>>(ex, p96, ord0, cat, cpos, rel6);
  gemmS(cat, F(50), F(51), poolo, nullptr, nullptr, 0, cG*48, 128, 134);
  k_ln<128><<<(cG*48+7)/8, 256, 0, stream>>>(poolo, F(52), F(53), xc, cG*48);
  k_morton<48,64><<<dim3(cG,2), 64, 0, stream>>>(cpos, ord1, inv1, rot1, irot1);
  runBlock1(54, ord1);
  runBlock1(71, rot1);
  k_unpool_prep<<<(cG*cNC+255)/256, 256, 0, stream>>>(xc, rel6, cat);
  gemmS(cat, F(88), F(89), unp, nullptr, nullptr, 0, cG*48, 128, 134);
  k_unpool_combine<<<(cG*96+7)/8, 256, 0, stream>>>(skip, unp, inv0, F(90), F(91), ex);
  runBlock0(92, ord0);
  runBlock0(109, rot0);

  // ---- stage 3: scatter back + output projection ----
  k_outein<<<4096, 256, 0, stream>>>(sw_tr, ex, out_hd);
  k_gemm_out<<<2048, 256, 0, stream>>>(out_hd, Wout_t, F(13), (float*)d_out);
}
